// TritonShortSeqSelfAttentionModule_53755810677033
// MI455X (gfx1250) — matrix operand order repaired and checked by reading, not compiled, not run on hardware
//
#include <hip/hip_runtime.h>

// ---------------------------------------------------------------------------
// Fused short-seq multi-head self-attention for MI455X (gfx1250, wave32).
//   x:[B,32,128] f32  wq/wk/wv:[4,128,32] f32  wo:[4,32,128] f32 -> out:[B,32,128] f32
// One wave per batch element; all GEMMs via v_wmma_f32_16x16x32_f16.
// Weights converted once per block to f16 and kept in LDS in B-fragment-
// friendly [n][k] layout. Per-wave LDS scratch performs the D->A layout
// shuffles between chained WMMAs (QK^T, PV, out-proj).
// Round-1 fix: prefetch all B fragments of each GEMM into registers before
// the WMMA burst (kills the per-WMMA s_wait_dscnt 0 stalls), and prefetch
// the next batch's x tile into cache (global_prefetch_b8).
// ---------------------------------------------------------------------------

typedef __attribute__((ext_vector_type(16))) _Float16 v16h;
typedef __attribute__((ext_vector_type(8)))  _Float16 v8h;
typedef __attribute__((ext_vector_type(8)))  float    v8f;

union V16 { v16h v; v8h h[2]; _Float16 e[16]; };

#define WAVES_PER_BLOCK 8
#define THREADS (WAVES_PER_BLOCK * 32)
#define NBLOCKS 1024
// LDS: 4 weight matrices (16384 f16 each) + 8 waves * 8192 f16 scratch
#define SHMEM_HALF (4 * 16384 + WAVES_PER_BLOCK * 8192)
#define SHMEM_BYTES (SHMEM_HALF * 2)

#define WMMA(a, b, c) \
  __builtin_amdgcn_wmma_f32_16x16x32_f16(false, (a), false, (b), (short)0, (c), false, false)

// ---- fragment loaders -----------------------------------------------------

// B fragment (K=32 x N=16, f16) from LDS stored [n][k] row-major (stride in halves).
// Lane layout: n=lr (both halves); half 0 keeps k {0..7,16..23}, half 1 {8..15,24..31}.
__device__ __forceinline__ v16h ldsB(const _Float16* base, int stride, int kt, int nt,
                                     int lr, int hf) {
  const _Float16* p = base + (nt * 16 + lr) * stride + kt * 32 + hf * 8;
  V16 r;
  r.h[0] = *(const v8h*)(p);        // k offsets +0..7
  r.h[1] = *(const v8h*)(p + 16);   // k offsets +16..23
  return r.v;
}

// A fragment (M=16 x K=32, f16) from LDS stored [m][k] row-major.
// Lane layout: row = mt*16+lr (both halves); half 0 keeps k {0..7,16..23}, half 1 {8..15,24..31}.
__device__ __forceinline__ v16h ldsA(const _Float16* base, int stride, int mt, int kt,
                                     int lr, int hf) {
  const _Float16* p = base + (mt * 16 + lr) * stride + kt * 32 + hf * 8;
  V16 r;
  r.h[0] = *(const v8h*)(p);        // k offsets +0..7
  r.h[1] = *(const v8h*)(p + 16);   // k offsets +16..23
  return r.v;
}

// A fragment straight from global fp32 x (row stride 128 floats), converted to f16.
__device__ __forceinline__ v16h gA(const float* row0, int mt, int kt, int lr, int hf) {
  const float* p = row0 + (size_t)(mt * 16 + lr) * 128 + kt * 32 + hf * 8;
  float4 f0 = ((const float4*)p)[0];
  float4 f1 = ((const float4*)p)[1];
  float4 f2 = ((const float4*)(p + 16))[0];
  float4 f3 = ((const float4*)(p + 16))[1];
  V16 r;
  r.e[0]  = (_Float16)f0.x;  r.e[1]  = (_Float16)f0.y;
  r.e[2]  = (_Float16)f0.z;  r.e[3]  = (_Float16)f0.w;
  r.e[4]  = (_Float16)f1.x;  r.e[5]  = (_Float16)f1.y;
  r.e[6]  = (_Float16)f1.z;  r.e[7]  = (_Float16)f1.w;
  r.e[8]  = (_Float16)f2.x;  r.e[9]  = (_Float16)f2.y;
  r.e[10] = (_Float16)f2.z;  r.e[11] = (_Float16)f2.w;
  r.e[12] = (_Float16)f3.x;  r.e[13] = (_Float16)f3.y;
  r.e[14] = (_Float16)f3.z;  r.e[15] = (_Float16)f3.w;
  return r.v;
}

// Store D-layout f32 accumulator tile to LDS as f16, row-major [m][n].
// D layout: VGPR r holds (row = mt*16 + hf*8 + r, col = nt*16 + lr).
__device__ __forceinline__ void stD(_Float16* base, int stride, int mt, int nt,
                                    int lr, int hf, v8f acc, float scale) {
  int col = nt * 16 + lr;
#pragma unroll
  for (int r = 0; r < 8; ++r) {
    int row = mt * 16 + hf * 8 + r;
    base[row * stride + col] = (_Float16)(acc[r] * scale);
  }
}

// Store D tile transposed ([n][m]); 8 consecutive rows pack into one b128 store.
__device__ __forceinline__ void stDT(_Float16* base, int stride, int mt, int nt,
                                     int lr, int hf, v8f acc) {
  int col = nt * 16 + lr;
  v8h pk;
#pragma unroll
  for (int r = 0; r < 8; ++r) pk[r] = (_Float16)acc[r];
  *(v8h*)(base + col * stride + mt * 16 + hf * 8) = pk;
}

// ---- 16-lane row reductions (rows never cross the 16-lane half in wave32) --
__device__ __forceinline__ float rmax16(float v) {
#pragma unroll
  for (int m = 1; m < 16; m <<= 1) v = fmaxf(v, __shfl_xor(v, m, 32));
  return v;
}
__device__ __forceinline__ float rsum16(float v) {
#pragma unroll
  for (int m = 1; m < 16; m <<= 1) v += __shfl_xor(v, m, 32);
  return v;
}

// One 32x32 = x[32,128] @ W[128,32] projection with all B fragments
// prefetched into registers before the WMMA burst.
__device__ __forceinline__ void proj_gemm(const v16h (&xa)[2][4], const _Float16* w,
                                          int lr, int hf, v8f (&acc)[2][2]) {
  v16h bf[2][4];
#pragma unroll
  for (int nt = 0; nt < 2; ++nt)
#pragma unroll
    for (int kt = 0; kt < 4; ++kt) bf[nt][kt] = ldsB(w, 128, kt, nt, lr, hf);
#pragma unroll
  for (int nt = 0; nt < 2; ++nt)
#pragma unroll
    for (int kt = 0; kt < 4; ++kt)
#pragma unroll
      for (int mt = 0; mt < 2; ++mt)
        acc[mt][nt] = WMMA(xa[mt][kt], bf[nt][kt], acc[mt][nt]);
}

__global__ __launch_bounds__(THREADS, 1)
void attn_fused_wmma_kernel(const float* __restrict__ x,  const float* __restrict__ wq,
                            const float* __restrict__ wk, const float* __restrict__ wv,
                            const float* __restrict__ wo, float* __restrict__ out,
                            int nbatch) {
  extern __shared__ _Float16 smem[];
  _Float16* WQ = smem;            // [h][d][e]  (B layout: n=d, k=e) 16384 halves
  _Float16* WK = WQ + 16384;
  _Float16* WV = WK + 16384;
  _Float16* WO = WV + 16384;      // [o][h*32+d] (B layout: n=o, k=hd)

  const int tid  = threadIdx.x;
  const int wave = tid >> 5;
  const int lane = tid & 31;
  const int lr   = lane & 15;
  const int hf   = lane >> 4;

  _Float16* scratch = WO + 16384 + wave * 8192;
  _Float16* tQ  = scratch;          // 32x32 [s][d]; reused as P [s][t]
  _Float16* tK  = scratch + 1024;   // 32x32 [t][d]
  _Float16* tVT = scratch + 2048;   // 32x32 [d][t]
  _Float16* tCX = scratch + 3072;   // 32x128 [s][h*32+d]

  // --- one-time: convert + transpose weights into LDS --------------------
#pragma unroll 1
  for (int i = tid; i < 16384; i += THREADS) {
    int h = i >> 12, rem = i & 4095;
    int e = rem >> 5, d = rem & 31;             // wq/wk/wv are [h][e][d]
    int dst = ((h << 5) + d) * 128 + e;         // -> [h][d][e]
    WQ[dst] = (_Float16)wq[i];
    WK[dst] = (_Float16)wk[i];
    WV[dst] = (_Float16)wv[i];
    int d2 = rem >> 7, o = rem & 127;           // wo is [h][d][o]
    WO[o * 128 + (h << 5) + d2] = (_Float16)wo[i];  // -> [o][h*32+d]
  }
  __syncthreads();

  const float qscale = 0.17677669529663687f;  // 32^-0.5
  const int bstride = gridDim.x * WAVES_PER_BLOCK;

#pragma unroll 1
  for (int b = blockIdx.x * WAVES_PER_BLOCK + wave; b < nbatch; b += bstride) {
    const float* xb = x + (size_t)b * 4096;

    // Prefetch the next batch's x tile into cache while we compute this one.
    if (b + bstride < nbatch) {
      const char* nx = (const char*)(x + (size_t)(b + bstride) * 4096);
      // 32 lanes x 4 x 128B = 16KB tile coverage
      __builtin_prefetch(nx + lane * 512 + 0, 0, 3);
      __builtin_prefetch(nx + lane * 512 + 128, 0, 3);
      __builtin_prefetch(nx + lane * 512 + 256, 0, 3);
      __builtin_prefetch(nx + lane * 512 + 384, 0, 3);
    }

    // x tile as resident A fragments (reused by all 12 QKV GEMMs)
    v16h xa[2][4];
#pragma unroll
    for (int mt = 0; mt < 2; ++mt)
#pragma unroll
      for (int kt = 0; kt < 4; ++kt) xa[mt][kt] = gA(xb, mt, kt, lr, hf);

#pragma unroll 1
    for (int h = 0; h < 4; ++h) {
      // ---- Q = x @ Wq ---------------------------------------------------
      {
        v8f acc[2][2] = {};
        proj_gemm(xa, WQ + h * 4096, lr, hf, acc);
#pragma unroll
        for (int mt = 0; mt < 2; ++mt)
#pragma unroll
          for (int nt = 0; nt < 2; ++nt) stD(tQ, 32, mt, nt, lr, hf, acc[mt][nt], qscale);
      }
      // ---- K = x @ Wk ---------------------------------------------------
      {
        v8f acc[2][2] = {};
        proj_gemm(xa, WK + h * 4096, lr, hf, acc);
#pragma unroll
        for (int mt = 0; mt < 2; ++mt)
#pragma unroll
          for (int nt = 0; nt < 2; ++nt) stD(tK, 32, mt, nt, lr, hf, acc[mt][nt], 1.0f);
      }
      // ---- V = x @ Wv (stored transposed for the PV B operand) ----------
      {
        v8f acc[2][2] = {};
        proj_gemm(xa, WV + h * 4096, lr, hf, acc);
#pragma unroll
        for (int mt = 0; mt < 2; ++mt)
#pragma unroll
          for (int nt = 0; nt < 2; ++nt) stDT(tVT, 32, mt, nt, lr, hf, acc[mt][nt]);
      }
      asm volatile("s_wait_dscnt 0" ::: "memory");  // shuffle tiles visible

      // ---- scores = (Q*scale) K^T  (K-dim = d = 32, single k-step) ------
      v8f sc[2][2] = {};
      {
        v16h aq[2], bk[2];
#pragma unroll
        for (int mt = 0; mt < 2; ++mt) aq[mt] = ldsA(tQ, 32, mt, 0, lr, hf);
#pragma unroll
        for (int nt = 0; nt < 2; ++nt) bk[nt] = ldsB(tK, 32, 0, nt, lr, hf);
#pragma unroll
        for (int mt = 0; mt < 2; ++mt)
#pragma unroll
          for (int nt = 0; nt < 2; ++nt) sc[mt][nt] = WMMA(aq[mt], bk[nt], sc[mt][nt]);
      }

      // ---- softmax over t (row = mt*16 + hf*8 + r lives in one 16-lane half)
      float rinv[2][8];
#pragma unroll
      for (int mt = 0; mt < 2; ++mt)
#pragma unroll
        for (int r = 0; r < 8; ++r) {
          float v0 = sc[mt][0][r], v1 = sc[mt][1][r];
          float m = rmax16(fmaxf(v0, v1));
          v0 = __expf(v0 - m);
          v1 = __expf(v1 - m);
          float s = rsum16(v0 + v1);
          rinv[mt][r] = 1.0f / s;
          sc[mt][0][r] = v0;
          sc[mt][1][r] = v1;
        }

      // ---- P tile to LDS (reuse tQ), then ctx = P @ V -------------------
#pragma unroll
      for (int mt = 0; mt < 2; ++mt)
#pragma unroll
        for (int nt = 0; nt < 2; ++nt) stD(tQ, 32, mt, nt, lr, hf, sc[mt][nt], 1.0f);
      asm volatile("s_wait_dscnt 0" ::: "memory");

      v8f cx[2][2] = {};
      {
        v16h ap[2], bv[2];
#pragma unroll
        for (int mt = 0; mt < 2; ++mt) ap[mt] = ldsA(tQ, 32, mt, 0, lr, hf);
#pragma unroll
        for (int nt = 0; nt < 2; ++nt) bv[nt] = ldsB(tVT, 32, 0, nt, lr, hf);
#pragma unroll
        for (int mt = 0; mt < 2; ++mt)
#pragma unroll
          for (int nt = 0; nt < 2; ++nt) cx[mt][nt] = WMMA(ap[mt], bv[nt], cx[mt][nt]);
      }

      // ---- normalized ctx into the concat buffer [s][h*32+d] ------------
#pragma unroll
      for (int mt = 0; mt < 2; ++mt)
#pragma unroll
        for (int nt = 0; nt < 2; ++nt)
#pragma unroll
          for (int r = 0; r < 8; ++r) {
            int row = mt * 16 + hf * 8 + r;
            tCX[row * 128 + h * 32 + nt * 16 + lr] =
                (_Float16)(cx[mt][nt][r] * rinv[mt][r]);
          }
    }  // heads
    asm volatile("s_wait_dscnt 0" ::: "memory");

    // ---- out = ctx_concat[32,128] @ WO[128,128] --------------------------
    v8f oacc[2][8] = {};
#pragma unroll
    for (int kt = 0; kt < 4; ++kt) {
      v16h a0 = ldsA(tCX, 128, 0, kt, lr, hf);
      v16h a1 = ldsA(tCX, 128, 1, kt, lr, hf);
      v16h bw[8];
#pragma unroll
      for (int nt = 0; nt < 8; ++nt) bw[nt] = ldsB(WO, 128, kt, nt, lr, hf);
#pragma unroll
      for (int nt = 0; nt < 8; ++nt) {
        oacc[0][nt] = WMMA(a0, bw[nt], oacc[0][nt]);
        oacc[1][nt] = WMMA(a1, bw[nt], oacc[1][nt]);
      }
    }
    float* ob = out + (size_t)b * 4096;
#pragma unroll
    for (int mt = 0; mt < 2; ++mt)
#pragma unroll
      for (int nt = 0; nt < 8; ++nt)
#pragma unroll
        for (int r = 0; r < 8; ++r)
          ob[(mt * 16 + hf * 8 + r) * 128 + nt * 16 + lr] = oacc[mt][nt][r];
  }  // batch loop
}

extern "C" void kernel_launch(void* const* d_in, const int* in_sizes, int n_in,
                              void* d_out, int out_size, void* d_ws, size_t ws_size,
                              hipStream_t stream) {
  const float* x  = (const float*)d_in[0];
  const float* wq = (const float*)d_in[1];
  const float* wk = (const float*)d_in[2];
  const float* wv = (const float*)d_in[3];
  const float* wo = (const float*)d_in[4];
  float* out = (float*)d_out;
  int nbatch = in_sizes[0] / (32 * 128);

  (void)hipFuncSetAttribute((const void*)attn_fused_wmma_kernel,
                            hipFuncAttributeMaxDynamicSharedMemorySize,
                            (int)SHMEM_BYTES);
  attn_fused_wmma_kernel<<<dim3(NBLOCKS), dim3(THREADS), SHMEM_BYTES, stream>>>(
      x, wq, wk, wv, wo, out, nbatch);
}
